// CausalAttention_3478923509860
// MI455X (gfx1250) — matrix operand order repaired and checked by reading, not compiled, not run on hardware
//
#include <hip/hip_runtime.h>
#include <hip/hip_bf16.h>

// ---------------------------------------------------------------------------
// Causal attention block for MI455X (gfx1250, wave32, WMMA 16x16x32 bf16)
//   1) qkv = x @ w_qkv                    (bf16 WMMA, f32 accum -> bf16)
//   2) RoPE (xPos scaling) on q,k         (elementwise)
//   3) flash attention per head           (bf16 WMMA, online softmax)
//   4) out = attn @ w_out                 (bf16 WMMA, f32 accum -> f32)
// GEMMs: 128x64 block tile, 4 waves (32x64 each), double-buffered LDS.
// ---------------------------------------------------------------------------

typedef __attribute__((ext_vector_type(16))) __bf16 v16bf;
typedef __attribute__((ext_vector_type(8)))  __bf16 v8bf;
typedef __attribute__((ext_vector_type(8)))  float  v8f;

#define S_LEN   2048
#define D_MODEL 2048
#define NH      16
#define DQK     128
#define QKV_N   (3 * NH * DQK)      // 6144
#define SM_SCALE 0.08838834764831845f   // 1/sqrt(128)

// Build a 32-byte A/B fragment from two 16-byte aligned chunks.
static __device__ __forceinline__ v16bf ld2(const __bf16* p0, const __bf16* p1) {
  v8bf a = *(const v8bf*)p0;
  v8bf b = *(const v8bf*)p1;
  return __builtin_shufflevector(a, b, 0, 1, 2, 3, 4, 5, 6, 7,
                                        8, 9, 10, 11, 12, 13, 14, 15);
}

static __device__ __forceinline__ v8f wmma_bf16(v16bf a, v16bf b, v8f c) {
  return __builtin_amdgcn_wmma_f32_16x16x32_bf16(
      /*neg_a=*/false, a, /*neg_b=*/false, b,
      /*c_mod=*/(short)0, c, /*reuse_a=*/false, /*reuse_b=*/false);
}

// ---------------------------------------------------------------------------
// GEMM: A (f32, MxK) @ B (f32, KxN) -> C (bf16, MxN)
// 128 threads (4 waves), tile 128x64, K step 32, double-buffered LDS.
// ---------------------------------------------------------------------------
__global__ __launch_bounds__(128)
void gemm_f32f32_bf16(const float* __restrict__ A, const float* __restrict__ B,
                      __bf16* __restrict__ C, int M, int N, int K) {
  __shared__ __align__(16) __bf16 As[2][128][40];   // row stride 80B
  __shared__ __align__(16) __bf16 Bs[2][32][72];    // row stride 144B

  const int tid  = threadIdx.x;
  const int lane = tid & 31;
  const int wave = tid >> 5;
  const int lo   = lane & 15;
  const int hi   = lane >> 4;
  const int m0   = blockIdx.y * 128;
  const int n0   = blockIdx.x * 64;

  v8f acc[2][4] = {};
  float4 ra[8];   // A tile 128x32 f32: 1024 float4 / 128 thr = 8 each
  float4 rb[4];   // B tile  32x64 f32:  512 float4 / 128 thr = 4 each

  auto stage_load = [&](int k0) {
    #pragma unroll
    for (int i = 0; i < 8; ++i) {
      int idx = tid + 128 * i;
      int r = idx >> 3, c4 = (idx & 7) * 4;
      ra[i] = *(const float4*)(A + (size_t)(m0 + r) * K + k0 + c4);
    }
    #pragma unroll
    for (int i = 0; i < 4; ++i) {
      int idx = tid + 128 * i;
      int r = idx >> 4, c4 = (idx & 15) * 4;
      rb[i] = *(const float4*)(B + (size_t)(k0 + r) * N + n0 + c4);
    }
  };
  auto stage_store = [&](int b) {
    #pragma unroll
    for (int i = 0; i < 8; ++i) {
      int idx = tid + 128 * i;
      int r = idx >> 3, c4 = (idx & 7) * 4;
      As[b][r][c4 + 0] = (__bf16)ra[i].x; As[b][r][c4 + 1] = (__bf16)ra[i].y;
      As[b][r][c4 + 2] = (__bf16)ra[i].z; As[b][r][c4 + 3] = (__bf16)ra[i].w;
    }
    #pragma unroll
    for (int i = 0; i < 4; ++i) {
      int idx = tid + 128 * i;
      int r = idx >> 4, c4 = (idx & 15) * 4;
      Bs[b][r][c4 + 0] = (__bf16)rb[i].x; Bs[b][r][c4 + 1] = (__bf16)rb[i].y;
      Bs[b][r][c4 + 2] = (__bf16)rb[i].z; Bs[b][r][c4 + 3] = (__bf16)rb[i].w;
    }
  };

  stage_load(0);
  stage_store(0);
  int buf = 0;
  const int kb = hi * 8;

  for (int k0 = 0; k0 < K; k0 += 32) {
    const bool has_next = (k0 + 32 < K);
    __syncthreads();                       // LDS[buf] ready; LDS[buf^1] free
    if (has_next) stage_load(k0 + 32);     // issue next tile's global loads
    if (k0 + 64 < K)
      __builtin_prefetch(A + (size_t)(m0 + tid) * K + k0 + 64, 0, 1);

    // compute from LDS[buf]: 2 A frags, 4 B frags, 8 WMMA
    v16bf a0 = ld2(&As[buf][wave * 32 + lo][kb],      &As[buf][wave * 32 + lo][kb + 16]);
    v16bf a1 = ld2(&As[buf][wave * 32 + 16 + lo][kb], &As[buf][wave * 32 + 16 + lo][kb + 16]);
    #pragma unroll
    for (int j = 0; j < 4; ++j) {
      v16bf b;
      for (int e = 0; e < 16; ++e) b[e] = Bs[buf][kb + (e & 7) + ((e >> 3) << 4)][j * 16 + lo];
      acc[0][j] = wmma_bf16(a0, b, acc[0][j]);
      acc[1][j] = wmma_bf16(a1, b, acc[1][j]);
    }
    if (has_next) stage_store(buf ^ 1);
    buf ^= 1;
  }

  #pragma unroll
  for (int mi = 0; mi < 2; ++mi)
    #pragma unroll
    for (int j = 0; j < 4; ++j)
      #pragma unroll
      for (int r = 0; r < 8; ++r) {
        int row = m0 + wave * 32 + mi * 16 + r + 8 * hi;
        int col = n0 + j * 16 + lo;
        C[(size_t)row * N + col] = (__bf16)acc[mi][j][r];
      }
}

// ---------------------------------------------------------------------------
// GEMM: A (bf16, MxK) @ B (f32, KxN) -> C (f32, MxN); same structure.
// ---------------------------------------------------------------------------
__global__ __launch_bounds__(128)
void gemm_bf16f32_f32(const __bf16* __restrict__ A, const float* __restrict__ B,
                      float* __restrict__ C, int M, int N, int K) {
  __shared__ __align__(16) __bf16 As[2][128][40];
  __shared__ __align__(16) __bf16 Bs[2][32][72];

  const int tid  = threadIdx.x;
  const int lane = tid & 31;
  const int wave = tid >> 5;
  const int lo   = lane & 15;
  const int hi   = lane >> 4;
  const int m0   = blockIdx.y * 128;
  const int n0   = blockIdx.x * 64;

  v8f acc[2][4] = {};
  uint4  ra[4];   // A tile 128x32 bf16: 512 uint4 / 128 thr = 4 each
  float4 rb[4];

  auto stage_load = [&](int k0) {
    #pragma unroll
    for (int i = 0; i < 4; ++i) {
      int idx = tid + 128 * i;
      int r = idx >> 2, q = idx & 3;     // 4 uint4 per 32-elem row
      ra[i] = *(const uint4*)(A + (size_t)(m0 + r) * K + k0 + 8 * q);
    }
    #pragma unroll
    for (int i = 0; i < 4; ++i) {
      int idx = tid + 128 * i;
      int r = idx >> 4, c4 = (idx & 15) * 4;
      rb[i] = *(const float4*)(B + (size_t)(k0 + r) * N + n0 + c4);
    }
  };
  auto stage_store = [&](int b) {
    #pragma unroll
    for (int i = 0; i < 4; ++i) {
      int idx = tid + 128 * i;
      int r = idx >> 2, q = idx & 3;
      *(uint4*)&As[b][r][8 * q] = ra[i];
    }
    #pragma unroll
    for (int i = 0; i < 4; ++i) {
      int idx = tid + 128 * i;
      int r = idx >> 4, c4 = (idx & 15) * 4;
      Bs[b][r][c4 + 0] = (__bf16)rb[i].x; Bs[b][r][c4 + 1] = (__bf16)rb[i].y;
      Bs[b][r][c4 + 2] = (__bf16)rb[i].z; Bs[b][r][c4 + 3] = (__bf16)rb[i].w;
    }
  };

  stage_load(0);
  stage_store(0);
  int buf = 0;
  const int kb = hi * 8;

  for (int k0 = 0; k0 < K; k0 += 32) {
    const bool has_next = (k0 + 32 < K);
    __syncthreads();
    if (has_next) stage_load(k0 + 32);
    if (k0 + 64 < K)
      __builtin_prefetch(B + (size_t)(k0 + 64 + (tid & 31)) * N + n0 + (tid >> 5) * 16, 0, 1);

    v16bf a0 = ld2(&As[buf][wave * 32 + lo][kb],      &As[buf][wave * 32 + lo][kb + 16]);
    v16bf a1 = ld2(&As[buf][wave * 32 + 16 + lo][kb], &As[buf][wave * 32 + 16 + lo][kb + 16]);
    #pragma unroll
    for (int j = 0; j < 4; ++j) {
      v16bf b;
      for (int e = 0; e < 16; ++e) b[e] = Bs[buf][kb + (e & 7) + ((e >> 3) << 4)][j * 16 + lo];
      acc[0][j] = wmma_bf16(a0, b, acc[0][j]);
      acc[1][j] = wmma_bf16(a1, b, acc[1][j]);
    }
    if (has_next) stage_store(buf ^ 1);
    buf ^= 1;
  }

  #pragma unroll
  for (int mi = 0; mi < 2; ++mi)
    #pragma unroll
    for (int j = 0; j < 4; ++j)
      #pragma unroll
      for (int r = 0; r < 8; ++r) {
        int row = m0 + wave * 32 + mi * 16 + r + 8 * hi;
        int col = n0 + j * 16 + lo;
        C[(size_t)row * N + col] = acc[mi][j][r];
      }
}

// ---------------------------------------------------------------------------
// RoPE with xPos scaling; splits qkv [S][6144] into per-head Q/K/V [h][s][128]
// ---------------------------------------------------------------------------
__global__ __launch_bounds__(256)
void rope_split(const __bf16* __restrict__ qkv, __bf16* __restrict__ Qo,
                __bf16* __restrict__ Ko, __bf16* __restrict__ Vo) {
  int idx = blockIdx.x * blockDim.x + threadIdx.x;
  if (idx >= S_LEN * NH * DQK) return;
  int d = idx & (DQK - 1);
  int h = (idx / DQK) & (NH - 1);
  int s = idx / (DQK * NH);

  const __bf16* base = qkv + (size_t)s * QKV_N + h * (3 * DQK);
  float qv = (float)base[d];
  float kv = (float)base[DQK + d];
  float vv = (float)base[2 * DQK + d];
  float qo = qv, ko = kv;

  if (d < 32) {
    int i = d & 15;  // pos/scale index repeats after 16
    float inv_freq = __powf(10000.0f, -(float)(2 * i) / 32.0f);
    float pos = (float)s * inv_freq;
    float cs = __cosf(pos), sn = __sinf(pos);
    float bscale = (2.0f * (float)i + 12.8f) / 44.8f;   // (2i + 0.4*32)/(1.4*32)
    float power = ((float)s - (float)(S_LEN / 2)) / 512.0f;
    float scale = __powf(bscale, power);
    int dp = (d < 16) ? d + 16 : d - 16;
    float qp = (float)base[dp];
    float kp = (float)base[DQK + dp];
    float rq = (d < 16) ? -qp : qp;   // rotate_half
    float rk = (d < 16) ? -kp : kp;
    qo = (qv * cs + rq * sn) * scale;
    ko = (kv * cs + rk * sn) / scale;
  }

  size_t off = (size_t)h * S_LEN * DQK + (size_t)s * DQK + d;
  Qo[off] = (__bf16)qo;
  Ko[off] = (__bf16)ko;
  Vo[off] = (__bf16)vv;
}

// ---------------------------------------------------------------------------
// Flash attention, causal. One WG (4 waves) = 64 query rows of one head.
// Each wave: 16 queries, 32-key tiles, online softmax, O += P @ V.
// ---------------------------------------------------------------------------
__global__ __launch_bounds__(128)
void flash_attn(const __bf16* __restrict__ Q, const __bf16* __restrict__ Kk,
                const __bf16* __restrict__ V, __bf16* __restrict__ O) {
  __shared__ __align__(16) __bf16 Kt[DQK][32];     // transposed: [dim][key]
  __shared__ __align__(16) __bf16 Vs[32][DQK];     // [key][dim]
  __shared__ __align__(16) __bf16 Ps[4][16][32];   // per-wave P transpose buffer

  const int tid  = threadIdx.x;
  const int lane = tid & 31;
  const int wave = tid >> 5;
  const int lo   = lane & 15;
  const int hi   = lane >> 4;
  const int h    = blockIdx.y;
  const int qwg  = blockIdx.x * 64;       // first query row of this WG
  const int qb   = qwg + wave * 16;       // first query row of this wave
  const size_t hoff = (size_t)h * S_LEN * DQK;

  // Q fragments for this wave's 16 rows, held in registers for the whole loop
  v16bf qf[4];
  {
    const __bf16* qrow = Q + hoff + (size_t)(qb + lo) * DQK;
    #pragma unroll
    for (int kc = 0; kc < 4; ++kc) {
      int kbb = kc * 32 + hi * 8;
      qf[kc] = ld2(qrow + kbb, qrow + kbb + 16);
    }
  }

  v8f o[8] = {};          // O: 16x128 f32 accum
  float mrow[8], lrow[8]; // per-row running max / sum (rows r + 8*hi)
  #pragma unroll
  for (int r = 0; r < 8; ++r) { mrow[r] = -1e30f; lrow[r] = 0.0f; }

  const int kend = qwg + 64;  // exclusive; all waves share the same trip count
  for (int kb0 = 0; kb0 < kend; kb0 += 32) {
    // batched K/V staging: 32 keys x 128 dims bf16 = 1024 uint2 / 128 thr = 8
    uint2 kr[8], vr[8];
    #pragma unroll
    for (int i = 0; i < 8; ++i) {
      int idx = tid + 128 * i;
      int key = idx >> 5, du2 = idx & 31;
      kr[i] = *(const uint2*)(Kk + hoff + (size_t)(kb0 + key) * DQK + 4 * du2);
      vr[i] = *(const uint2*)(V  + hoff + (size_t)(kb0 + key) * DQK + 4 * du2);
    }
    __syncthreads();
    #pragma unroll
    for (int i = 0; i < 8; ++i) {
      int idx = tid + 128 * i;
      int key = idx >> 5, du2 = idx & 31;
      *(uint2*)&Vs[key][4 * du2] = vr[i];
      const __bf16* kp = (const __bf16*)&kr[i];
      Kt[4 * du2 + 0][key] = kp[0];
      Kt[4 * du2 + 1][key] = kp[1];
      Kt[4 * du2 + 2][key] = kp[2];
      Kt[4 * du2 + 3][key] = kp[3];
    }
    __syncthreads();

    // scores S = Q @ K^T (16 x 32), two 16x16 fragments
    v8f s[2] = {};
    #pragma unroll
    for (int kc = 0; kc < 4; ++kc) {
      #pragma unroll
      for (int j = 0; j < 2; ++j) {
        v16bf b;
        for (int e = 0; e < 16; ++e) b[e] = Kt[kc * 32 + hi * 8 + (e & 7) + ((e >> 3) << 4)][j * 16 + lo];
        s[j] = wmma_bf16(qf[kc], b, s[j]);
      }
    }

    // scale + causal mask + row max
    float mnew[8];
    #pragma unroll
    for (int r = 0; r < 8; ++r) mnew[r] = mrow[r];
    #pragma unroll
    for (int j = 0; j < 2; ++j)
      #pragma unroll
      for (int r = 0; r < 8; ++r) {
        int qr = qb + r + 8 * hi;
        int kc = kb0 + j * 16 + lo;
        float sv = s[j][r] * SM_SCALE;
        sv = (kc <= qr) ? sv : -1e30f;
        s[j][r] = sv;
        mnew[r] = fmaxf(mnew[r], sv);
      }
    #pragma unroll
    for (int msk = 1; msk < 16; msk <<= 1)
      #pragma unroll
      for (int r = 0; r < 8; ++r)
        mnew[r] = fmaxf(mnew[r], __shfl_xor(mnew[r], msk, 32));

    float alpha[8];
    #pragma unroll
    for (int r = 0; r < 8; ++r) {
      alpha[r] = __expf(mrow[r] - mnew[r]);
      mrow[r] = mnew[r];
    }

    // p = exp(s - m); transpose into A-fragment layout via per-wave LDS
    float psum[8];
    #pragma unroll
    for (int r = 0; r < 8; ++r) psum[r] = 0.0f;
    #pragma unroll
    for (int j = 0; j < 2; ++j)
      #pragma unroll
      for (int r = 0; r < 8; ++r) {
        float p = __expf(s[j][r] - mrow[r]);
        psum[r] += p;
        Ps[wave][r + 8 * hi][j * 16 + lo] = (__bf16)p;
      }
    #pragma unroll
    for (int msk = 1; msk < 16; msk <<= 1)
      #pragma unroll
      for (int r = 0; r < 8; ++r)
        psum[r] += __shfl_xor(psum[r], msk, 32);
    #pragma unroll
    for (int r = 0; r < 8; ++r) lrow[r] = lrow[r] * alpha[r] + psum[r];

    // rescale O by alpha
    #pragma unroll
    for (int c = 0; c < 8; ++c)
      #pragma unroll
      for (int r = 0; r < 8; ++r) o[c][r] *= alpha[r];

    // P A-fragment (16 rows x 32 keys)
    v16bf pa;
    {
      int kbb = hi * 8;
      pa = ld2(&Ps[wave][lo][kbb], &Ps[wave][lo][kbb + 16]);
    }
    // O += P @ V, 8 chunks of 16 output dims
    #pragma unroll
    for (int c = 0; c < 8; ++c) {
      v16bf vb;
      for (int e = 0; e < 16; ++e) vb[e] = Vs[hi * 8 + (e & 7) + ((e >> 3) << 4)][c * 16 + lo];
      o[c] = wmma_bf16(pa, vb, o[c]);
    }
  }

  // epilogue: normalize and write [s][h*128 + d] bf16
  #pragma unroll
  for (int c = 0; c < 8; ++c)
    #pragma unroll
    for (int r = 0; r < 8; ++r) {
      int row = qb + r + 8 * hi;
      int col = h * DQK + c * 16 + lo;
      O[(size_t)row * (NH * DQK) + col] = (__bf16)(o[c][r] / lrow[r]);
    }
}

// ---------------------------------------------------------------------------
extern "C" void kernel_launch(void* const* d_in, const int* in_sizes, int n_in,
                              void* d_out, int out_size, void* d_ws, size_t ws_size,
                              hipStream_t stream) {
  const float* x     = (const float*)d_in[0];
  const float* w_qkv = (const float*)d_in[1];
  const float* w_out = (const float*)d_in[2];
  float* out = (float*)d_out;

  char* ws = (char*)d_ws;
  const size_t qkv_elems  = (size_t)S_LEN * QKV_N;        // 12.58M bf16
  const size_t head_elems = (size_t)NH * S_LEN * DQK;     // 4.19M bf16 each

  __bf16* qkv  = (__bf16*)ws;
  __bf16* Qb   = qkv + qkv_elems;
  __bf16* Kb   = Qb + head_elems;
  __bf16* Vb   = Kb + head_elems;
  __bf16* attn = Vb + head_elems;   // total ~59 MB of workspace

  // 1) qkv = x @ w_qkv : M=2048, N=6144, K=2048
  gemm_f32f32_bf16<<<dim3(QKV_N / 64, S_LEN / 128), 128, 0, stream>>>(
      x, w_qkv, qkv, S_LEN, QKV_N, D_MODEL);

  // 2) RoPE + split into per-head Q/K/V
  rope_split<<<(S_LEN * NH * DQK) / 256, 256, 0, stream>>>(qkv, Qb, Kb, Vb);

  // 3) causal flash attention: 32 q-blocks x 16 heads
  flash_attn<<<dim3(S_LEN / 64, NH), 128, 0, stream>>>(Qb, Kb, Vb, attn);

  // 4) out = attn @ w_out : M=N=K=2048
  gemm_bf16f32_f32<<<dim3(D_MODEL / 64, S_LEN / 128), 128, 0, stream>>>(
      attn, w_out, out, S_LEN, D_MODEL, NH * DQK);
}
